// PA_MOE_40415642255781
// MI455X (gfx1250) — compile-verified, matrix operand order checked by reading, not run on hardware
//
#include <hip/hip_runtime.h>
#include <hip/hip_bf16.h>
#include <math.h>

// ---------------------------------------------------------------------------
// PA-MoE on MI455X (gfx1250): compute-bound (≈7.7e11 FLOPs vs 268MB traffic),
// so the 4 expert convs run as implicit GEMM on v_wmma_f32_16x16x32_bf16.
// Round 2: double-buffered weight staging (1 barrier/tap, global latency hidden
// behind WMMA work) + swizzle moved into the one-time pack kernel.
// ---------------------------------------------------------------------------

typedef __attribute__((ext_vector_type(16))) __bf16 v16bf;
typedef __attribute__((ext_vector_type(8)))  float  v8f;
typedef __attribute__((ext_vector_type(4)))  int    v4i;

union Frag { v4i q[2]; v16bf v; };

__device__ __forceinline__ unsigned short f2bf(float f) {
    unsigned u = __float_as_uint(f);
    u += 0x7FFFu + ((u >> 16) & 1u);          // round-to-nearest-even
    return (unsigned short)(u >> 16);
}

// Rows are 128B (64 bf16 channels). 16B granule g of row r is XOR-swizzled so
// that 16 lanes reading the same granule column from 16 consecutive rows hit
// all 64 LDS banks once (256B period = 2 rows, (r%2,(r>>1)&7) enumerates slots).
__device__ __forceinline__ int swz(int row, int g) {
    return row * 128 + ((g ^ ((row >> 1) & 7)) << 4);
}

// ------------------------- gating network ---------------------------------

__global__ void mean_kernel(const float* __restrict__ x, float* __restrict__ means) {
    __shared__ float red[256];
    const int bc = blockIdx.x;                    // b*64 + c
    const float* p = x + ((size_t)bc << 16);      // 256*256 plane
    float s = 0.f;
    for (int i = threadIdx.x; i < 65536; i += 256) s += p[i];
    red[threadIdx.x] = s;
    __syncthreads();
    for (int st = 128; st > 0; st >>= 1) {
        if (threadIdx.x < st) red[threadIdx.x] += red[threadIdx.x + st];
        __syncthreads();
    }
    if (threadIdx.x == 0) means[bc] = red[0] * (1.0f / 65536.0f);
}

__global__ void gate_kernel(const float* __restrict__ means,
                            const float* __restrict__ gw1, const float* __restrict__ gb1,
                            const float* __restrict__ gw2, const float* __restrict__ gb2,
                            float* __restrict__ coef, float* __restrict__ loss_out) {
    __shared__ float hid[8 * 512];
    __shared__ float lg[32];
    const int t = threadIdx.x;
    for (int i = t; i < 8 * 512; i += 256) {
        int b = i >> 9, h = i & 511;
        const float* w = gw1 + h * 64;
        const float* m = means + b * 64;
        float s = gb1[h];
        #pragma unroll 8
        for (int c = 0; c < 64; ++c) s += m[c] * w[c];
        hid[i] = fmaxf(s, 0.f);
    }
    __syncthreads();
    if (t < 32) {
        int b = t >> 2, e = t & 3;
        const float* w = gw2 + e * 512;
        const float* hm = hid + b * 512;
        float s = gb2[e];
        for (int h = 0; h < 512; ++h) s += hm[h] * w[h];
        lg[t] = s;
    }
    __syncthreads();
    if (t == 0) {
        float prob[32];
        for (int b = 0; b < 8; ++b) {
            float mx = lg[b * 4];
            for (int e = 1; e < 4; ++e) mx = fmaxf(mx, lg[b * 4 + e]);
            float den = 0.f;
            for (int e = 0; e < 4; ++e) { prob[b*4+e] = expf(lg[b*4+e] - mx); den += prob[b*4+e]; }
            float inv = 1.f / den;
            for (int e = 0; e < 4; ++e) prob[b*4+e] *= inv;
        }
        float mu = 0.f;
        for (int i = 0; i < 32; ++i) mu += prob[i];
        mu *= (1.f / 32.f);
        float var = 0.f;
        for (int i = 0; i < 32; ++i) { float d = prob[i] - mu; var += d * d; }
        var *= (1.f / 31.f);                       // ddof=1
        float loss = sqrtf(var) / (mu + 1e-10f) * 0.1f;
        float cf[4] = {0.f, 0.f, 0.f, 0.f};
        for (int b = 0; b < 8; ++b) {
            int i0 = 0;
            for (int e = 1; e < 4; ++e) if (prob[b*4+e] > prob[b*4+i0]) i0 = e;
            int i1 = -1;
            for (int e = 0; e < 4; ++e) {
                if (e == i0) continue;
                if (i1 < 0 || prob[b*4+e] > prob[b*4+i1]) i1 = e;
            }
            float e1 = expf(prob[b*4+i1] - prob[b*4+i0]);   // softmax over top-2 values
            float s0 = 1.f / (1.f + e1);
            cf[i0] += s0;
            cf[i1] += 1.f - s0;
        }
        for (int e = 0; e < 4; ++e) coef[e] = cf[e];
        *loss_out = loss;
    }
}

// --- weight pack: fp32 [co][ci][ky][kx] -> bf16 [tap][swizzled co x ci slice]
// The LDS bank swizzle is applied HERE (one-time), so the hot loop stages the
// 8KB slice with pure linear 16B copies.

__global__ void pack_kernel(const float* __restrict__ ew1, const float* __restrict__ ew3,
                            const float* __restrict__ ew7, const float* __restrict__ ew11,
                            unsigned short* __restrict__ packed) {
    int idx = blockIdx.x * 256 + threadIdx.x;     // 180 taps * 64 * 64
    if (idx >= 180 * 4096) return;
    int p  = idx >> 12;
    int co = (idx >> 6) & 63;
    int ci = idx & 63;
    const float* w; int k, pos;
    if      (p < 1)  { w = ew1;  k = 1;  pos = p;      }
    else if (p < 10) { w = ew3;  k = 3;  pos = p - 1;  }
    else if (p < 59) { w = ew7;  k = 7;  pos = p - 10; }
    else             { w = ew11; k = 11; pos = p - 59; }
    int ky = pos / k, kx = pos - ky * k;
    float v = w[((co * 64 + ci) * k + ky) * k + kx];
    int g = ci >> 3, sub = ci & 7;
    int dst = p * 4096 + (swz(co, g) >> 1) + sub; // ushort units, pre-swizzled
    packed[dst] = f2bf(v);
}

// ------------------------- fused MoE conv ----------------------------------
// Grid: (16,16,8) blocks x 256 threads (8 waves). One 16x16 output tile/block.
// Implicit GEMM per (dy,dx) tap: Y[co,pix] += W[co,:] * X[:,pix], K=64 (2x32).
// A-frag = weights (M=co), B-frag = x pixels (N=pix) so f32 D stores are
// contiguous along w. Weight slices double-buffered in LDS.

__launch_bounds__(256, 1)
__global__ void moe_conv_kernel(const float* __restrict__ x,
                                const unsigned short* __restrict__ packed,
                                const float* __restrict__ coef,
                                const float* __restrict__ eb1, const float* __restrict__ eb3,
                                const float* __restrict__ eb7, const float* __restrict__ eb11,
                                float* __restrict__ out) {
    __shared__ __align__(16) unsigned char s_x[676 * 128];      // 26x26 halo tile, bf16
    __shared__ __align__(16) unsigned char s_w[2][64 * 128];    // double-buffered 8KB slices

    const int tx0  = blockIdx.x * 16;
    const int ty0  = blockIdx.y * 16;
    const int b    = blockIdx.z;
    const int t    = threadIdx.x;
    const int lane = t & 31;
    const int wave = t >> 5;
    const int ln   = lane & 15;   // N index (pixel x) / M row for A
    const int hl   = lane >> 4;   // lane half selects K-granule set

    // ---- stage x tile (halo 5) as bf16, swizzled [pixel][channel] rows ----
    for (int c = 0; c < 64; ++c) {
        const float* xc = x + ((size_t)(b * 64 + c) << 16);
        const int gran = c >> 3, sub = c & 7;
        for (int p = t; p < 676; p += 256) {
            int hy = p / 26, hx = p - hy * 26;
            int gy = ty0 + hy - 5, gx = tx0 + hx - 5;
            float v = 0.f;
            if (gy >= 0 && gy < 256 && gx >= 0 && gx < 256) v = xc[(gy << 8) + gx];
            *(unsigned short*)(s_x + swz(p, gran) + sub * 2) = f2bf(v);
        }
    }

    const float cf0 = coef[0], cf1 = coef[1], cf2 = coef[2], cf3 = coef[3];
    const float cfs[4] = {cf0, cf1, cf2, cf3};
    const float* ebs[4] = {eb1, eb3, eb7, eb11};
    const int ks[4]   = {1, 3, 7, 11};
    const int poff[4] = {0, 1, 10, 59};
    const unsigned char* packedb = (const unsigned char*)packed;

    const v8f vzero = {0.f, 0.f, 0.f, 0.f, 0.f, 0.f, 0.f, 0.f};
    v8f outw[2][4];
    #pragma unroll
    for (int r = 0; r < 2; ++r)
        #pragma unroll
        for (int ct = 0; ct < 4; ++ct) outw[r][ct] = vzero;

    const int rowy0 = wave * 2;   // two output rows per wave

    for (int e = 0; e < 4; ++e) {
        const int k    = ks[e];
        const int off  = 5 - (k >> 1);
        const int npos = k * k;
        v8f acc[2][4];
        #pragma unroll
        for (int r = 0; r < 2; ++r)
            #pragma unroll
            for (int ct = 0; ct < 4; ++ct) acc[r][ct] = vzero;

        // prologue: stage tap 0 into buffer 0 (linear copy, pre-swizzled data)
        {
            const v4i* wp = (const v4i*)(packedb + (size_t)poff[e] * 8192);
            __syncthreads();                       // buffers may be in use by prev expert
            *(v4i*)(s_w[0] + t * 16)         = wp[t];
            *(v4i*)(s_w[0] + (t + 256) * 16) = wp[t + 256];
            __syncthreads();
        }

        int dy = 0, dx = 0;
        for (int pos = 0; pos < npos; ++pos) {
            const int cur = pos & 1;
            const bool have_next = (pos + 1 < npos);

            // issue next tap's global loads early; latency hidden by WMMAs below
            v4i nw0, nw1;
            if (have_next) {
                const v4i* wp = (const v4i*)(packedb + (size_t)(poff[e] + pos + 1) * 8192);
                nw0 = wp[t];
                nw1 = wp[t + 256];
            }

            const unsigned char* swc = s_w[cur];

            // A-frags: weights, M=co (row = ct*16+ln), K = ci chunk cc*32
            Frag afr[4][2];
            #pragma unroll
            for (int ct = 0; ct < 4; ++ct)
                #pragma unroll
                for (int cc = 0; cc < 2; ++cc) {
                    int row = ct * 16 + ln;
                    int g0 = cc * 4 + hl;
                    afr[ct][cc].q[0] = *(const v4i*)(swc + swz(row, g0));
                    afr[ct][cc].q[1] = *(const v4i*)(swc + swz(row, g0 + 2));
                }
            // B-frags: x pixels, N=pixel (column = output x = tx0+ln)
            Frag bfr[2][2];
            #pragma unroll
            for (int r = 0; r < 2; ++r) {
                int hy  = rowy0 + r + dy + off;
                int row = hy * 26 + dx + off + ln;
                #pragma unroll
                for (int cc = 0; cc < 2; ++cc) {
                    int g0 = cc * 4 + hl;
                    bfr[r][cc].q[0] = *(const v4i*)(s_x + swz(row, g0));
                    bfr[r][cc].q[1] = *(const v4i*)(s_x + swz(row, g0 + 2));
                }
            }
            #pragma unroll
            for (int r = 0; r < 2; ++r)
                #pragma unroll
                for (int ct = 0; ct < 4; ++ct)
                    #pragma unroll
                    for (int cc = 0; cc < 2; ++cc)
                        acc[r][ct] = __builtin_amdgcn_wmma_f32_16x16x32_bf16(
                            false, afr[ct][cc].v, false, bfr[r][cc].v,
                            (short)0, acc[r][ct], false, false);

            // write next tap into the other buffer, then one barrier per tap
            if (have_next) {
                unsigned char* dst = s_w[cur ^ 1];
                *(v4i*)(dst + t * 16)         = nw0;
                *(v4i*)(dst + (t + 256) * 16) = nw1;
            }
            __syncthreads();

            if (++dx == k) { dx = 0; ++dy; }
        }

        // bias + sigmoid + gate-weighted accumulate
        const float* eb = ebs[e];
        const float ce = cfs[e];
        #pragma unroll
        for (int ct = 0; ct < 4; ++ct)
            #pragma unroll
            for (int g = 0; g < 8; ++g) {
                const int co = ct * 16 + hl * 8 + g;
                const float bia = eb[co];
                #pragma unroll
                for (int r = 0; r < 2; ++r) {
                    float v = acc[r][ct][g] + bia;
                    outw[r][ct][g] += ce * (1.f / (1.f + __expf(-v)));
                }
            }
    }

    // out = x * sum_e coef_e*sigmoid(conv_e)
    #pragma unroll
    for (int r = 0; r < 2; ++r) {
        const int y = ty0 + rowy0 + r;
        #pragma unroll
        for (int ct = 0; ct < 4; ++ct)
            #pragma unroll
            for (int g = 0; g < 8; ++g) {
                const int co = ct * 16 + hl * 8 + g;
                const size_t idx = (((size_t)(b * 64 + co) << 8 | (unsigned)y) << 8) + tx0 + ln;
                out[idx] = x[idx] * outw[r][ct][g];
            }
    }
}

// ---------------------------------------------------------------------------

extern "C" void kernel_launch(void* const* d_in, const int* in_sizes, int n_in,
                              void* d_out, int out_size, void* d_ws, size_t ws_size,
                              hipStream_t stream) {
    const float* x    = (const float*)d_in[0];
    const float* gw1  = (const float*)d_in[1];
    const float* gb1  = (const float*)d_in[2];
    const float* gw2  = (const float*)d_in[3];
    const float* gb2  = (const float*)d_in[4];
    const float* ew1  = (const float*)d_in[5];
    const float* eb1  = (const float*)d_in[6];
    const float* ew3  = (const float*)d_in[7];
    const float* eb3  = (const float*)d_in[8];
    const float* ew7  = (const float*)d_in[9];
    const float* eb7  = (const float*)d_in[10];
    const float* ew11 = (const float*)d_in[11];
    const float* eb11 = (const float*)d_in[12];
    float* out = (float*)d_out;

    float* coef            = (float*)d_ws;                         // 4 floats
    float* means           = (float*)((char*)d_ws + 64);           // 512 floats
    unsigned short* packed = (unsigned short*)((char*)d_ws + 4096);// 180*4096 bf16 = 1.44MB

    mean_kernel<<<512, 256, 0, stream>>>(x, means);
    gate_kernel<<<1, 256, 0, stream>>>(means, gw1, gb1, gw2, gb2, coef, out + (out_size - 1));
    pack_kernel<<<2880, 256, 0, stream>>>(ew1, ew3, ew7, ew11, packed);

    dim3 grid(16, 16, 8);
    moe_conv_kernel<<<grid, 256, 0, stream>>>(x, packed, coef, eb1, eb3, eb7, eb11, out);
}
